// MultiHeadAttention_34187939676168
// MI455X (gfx1250) — compile-verified, matrix operand order checked by reading, not run on hardware
//
#include <hip/hip_runtime.h>
#include <hip/hip_bf16.h>

typedef __bf16 bf16_t;
typedef __attribute__((ext_vector_type(16))) __bf16 v16bf;
typedef __attribute__((ext_vector_type(8)))  __bf16 v8bf;
typedef __attribute__((ext_vector_type(8)))  float  v8f;

#define L_SEQ   2048
#define D_MODEL 1024
#define N_HEADS 16
#define D_HEAD  64
#define BATCH   2
#define NEGVAL  (-10000.0f)

// D = A(16x32 bf16) * B(32x16 bf16) + C(16x16 f32)
__device__ __forceinline__ v8f wmma_bf16(v16bf a, v16bf b, v8f c) {
  return __builtin_amdgcn_wmma_f32_16x16x32_bf16(false, a, false, b,
                                                 (short)0, c, false, false);
}

// 16-bit A/B fragment loader for a row-major source (row stride given by
// caller via pointer math). Lane layout (ISA 7.12.2): lanes 0-15 hold
// K {0..7, 16..23}, lanes 16-31 hold K {8..15, 24..31} -> two contiguous
// 16B chunks per lane.
__device__ __forceinline__ v16bf load_frag(const bf16_t* row, int hi) {
  v8bf lo = *(const v8bf*)(row + 8 * hi);
  v8bf up = *(const v8bf*)(row + 16 + 8 * hi);
  v16bf f;
#pragma unroll
  for (int i = 0; i < 8; ++i) { f[i] = lo[i]; f[8 + i] = up[i]; }
  return f;
}

// ASYNCcnt-tracked global->LDS copy, 16 bytes per lane (no VGPR round-trip).
// LDS address is taken from the vdst VGPR; generic shared pointers carry the LDS offset in
// their low 32 bits (ISA 10.2 aperture rules).
__device__ __forceinline__ void async_copy_b128(const bf16_t* gptr, bf16_t* lptr) {
  const unsigned int lds = (unsigned int)(unsigned long long)lptr;
  const unsigned long long ga = (unsigned long long)gptr;
  asm volatile("global_load_async_to_lds_b128 %0, %1, off"
               :: "v"(lds), "v"(ga) : "memory");
}

__device__ __forceinline__ void wait_async0() {
  asm volatile("s_wait_asynccnt 0" ::: "memory");
}

__global__ __launch_bounds__(256) void cvt_f32_to_bf16(
    const float* __restrict__ in, bf16_t* __restrict__ out, int n) {
  int i = blockIdx.x * blockDim.x + threadIdx.x;
  const int stride = gridDim.x * blockDim.x;
  for (; i < n; i += stride) out[i] = (bf16_t)in[i];
}

// C[M,N] = A[M,K] * W[N,K]^T + bias, bf16 inputs, f32 accumulate.
// Block = 256 threads = 8 waves; block tile 128x128; wave tile 32x64.
// A/B panels double-buffered in LDS via async global->LDS copies.
// mode 0: bf16 out, [B,H,L,dk] layout, scaled (Q/K projections)
// mode 1: bf16 out, [B,H,dk,L] layout (V transposed for the PV matmul)
// mode 2: f32 out, row-major [M,N] (final projection -> d_out)
__global__ __launch_bounds__(256) void gemm_bf16_wmma(
    const bf16_t* __restrict__ A, const bf16_t* __restrict__ W,
    const float* __restrict__ bias, void* __restrict__ outv,
    int M, int N, int K, int mode, float scale) {
  __shared__ __align__(16) bf16_t Abuf[2][128][32];
  __shared__ __align__(16) bf16_t Bbuf[2][128][32];

  const int tid  = threadIdx.x;
  const int wave = tid >> 5;
  const int lane = tid & 31;
  const int lr = lane & 15, hi = lane >> 4;
  const int wm = wave >> 1;            // 0..3 -> 32-row sub-tile
  const int wn = wave & 1;             // 0..1 -> 64-col sub-tile
  const int nblk = N >> 7;
  const int bm = blockIdx.x / nblk;
  const int bn = blockIdx.x % nblk;

  // staging map: 2 threads per row, 32B (2 x b128) per thread
  const int srow = tid >> 1, shalf = tid & 1;
  const bf16_t* ag = A + (size_t)(bm * 128 + srow) * K + shalf * 16;
  const bf16_t* wg = W + (size_t)(bn * 128 + srow) * K + shalf * 16;

  const v8f zero = {};
  v8f c[2][4];
#pragma unroll
  for (int i = 0; i < 2; ++i)
#pragma unroll
    for (int j = 0; j < 4; ++j) c[i][j] = zero;

  // prologue: stage k-tile 0 into buffer 0
  async_copy_b128(ag, &Abuf[0][srow][shalf * 16]);
  async_copy_b128(ag + 8, &Abuf[0][srow][shalf * 16 + 8]);
  async_copy_b128(wg, &Bbuf[0][srow][shalf * 16]);
  async_copy_b128(wg + 8, &Bbuf[0][srow][shalf * 16 + 8]);

  const int nk = K >> 5;
  for (int it = 0; it < nk; ++it) {
    const int buf = it & 1;
    wait_async0();        // my wave's async copies landed in LDS
    __syncthreads();      // whole tile visible; prior compute on buf^1 done

    if (it + 1 < nk) {    // stage next tile into the other buffer
      const int k1 = (it + 1) << 5;
      async_copy_b128(ag + k1, &Abuf[buf ^ 1][srow][shalf * 16]);
      async_copy_b128(ag + k1 + 8, &Abuf[buf ^ 1][srow][shalf * 16 + 8]);
      async_copy_b128(wg + k1, &Bbuf[buf ^ 1][srow][shalf * 16]);
      async_copy_b128(wg + k1 + 8, &Bbuf[buf ^ 1][srow][shalf * 16 + 8]);
    }

    v16bf a[2], b[4];
#pragma unroll
    for (int i = 0; i < 2; ++i)
      a[i] = load_frag(&Abuf[buf][wm * 32 + i * 16 + lr][0], hi);
#pragma unroll
    for (int j = 0; j < 4; ++j)
      b[j] = load_frag(&Bbuf[buf][wn * 64 + j * 16 + lr][0], hi);
#pragma unroll
    for (int i = 0; i < 2; ++i)
#pragma unroll
      for (int j = 0; j < 4; ++j)
        c[i][j] = wmma_bf16(a[i], b[j], c[i][j]);
  }

#pragma unroll
  for (int i = 0; i < 2; ++i) {
#pragma unroll
    for (int j = 0; j < 4; ++j) {
      const int nn = bn * 128 + wn * 64 + j * 16 + lr;   // C layout: N on lanes
      const float bb = bias[nn];
#pragma unroll
      for (int r = 0; r < 8; ++r) {                      // C layout: M = r + 8*hi
        const int mm = bm * 128 + wm * 32 + i * 16 + hi * 8 + r;
        const float v = (c[i][j][r] + bb) * scale;
        if (mode == 2) {
          ((float*)outv)[(size_t)mm * N + nn] = v;
        } else {
          const int b_ = mm >> 11, l = mm & (L_SEQ - 1);
          const int h = nn >> 6, d = nn & (D_HEAD - 1);
          bf16_t* o = (bf16_t*)outv;
          if (mode == 0)
            o[(((size_t)(b_ * N_HEADS + h)) * L_SEQ + l) * D_HEAD + d] = (bf16_t)v;
          else
            o[(((size_t)(b_ * N_HEADS + h)) * D_HEAD + d) * L_SEQ + l] = (bf16_t)v;
        }
      }
    }
  }
}

// Flash-attention: one wave = 16 query rows x full dk=64, streaming 32-key
// tiles. Q is pre-scaled by 1/sqrt(dk). V comes in transposed [dk, L].
__global__ __launch_bounds__(256) void attn_wmma(
    const bf16_t* __restrict__ Q, const bf16_t* __restrict__ K,
    const bf16_t* __restrict__ Vt, const int* __restrict__ mask,
    bf16_t* __restrict__ O) {
  __shared__ __align__(16) bf16_t Plds[8][16][32];  // per-wave P staging
  const int wave = threadIdx.x >> 5;
  const int lane = threadIdx.x & 31;
  const int lr = lane & 15, hi = lane >> 4;
  const int bh = blockIdx.y;                        // b*16 + h
  const int q0 = blockIdx.x * 128 + wave * 16;

  const bf16_t* Qh = Q  + (size_t)bh * L_SEQ * D_HEAD;
  const bf16_t* Kh = K  + (size_t)bh * L_SEQ * D_HEAD;
  const bf16_t* Vh = Vt + (size_t)bh * D_HEAD * L_SEQ;

  v16bf qa[2];
#pragma unroll
  for (int d0 = 0; d0 < 2; ++d0)
    qa[d0] = load_frag(Qh + (size_t)(q0 + lr) * D_HEAD + d0 * 32, hi);

  const v8f zero = {};
  float rm[8], rl[8];
  v8f o[4];
#pragma unroll
  for (int r = 0; r < 8; ++r) { rm[r] = -3.0e38f; rl[r] = 0.0f; }
#pragma unroll
  for (int j = 0; j < 4; ++j) o[j] = zero;

  const int* mrow = mask + (size_t)(q0 + hi * 8) * L_SEQ;

  for (int kb = 0; kb < L_SEQ; kb += 32) {
    // ---- S = (Q/sqrt(dk)) K^T : two 16-key column fragments ----
    v8f s[2]; s[0] = zero; s[1] = zero;
#pragma unroll
    for (int kh = 0; kh < 2; ++kh)
#pragma unroll
      for (int d0 = 0; d0 < 2; ++d0) {
        v16bf bk = load_frag(Kh + (size_t)(kb + kh * 16 + lr) * D_HEAD + d0 * 32, hi);
        s[kh] = wmma_bf16(qa[d0], bk, s[kh]);
      }

    // ---- mask + online softmax (N sits on lanes: reduce across 16-lane halves)
#pragma unroll
    for (int r = 0; r < 8; ++r) {
      const int* mr = mrow + (size_t)r * L_SEQ + kb;
      float v0 = (mr[lr] != 0)      ? s[0][r] : NEGVAL;
      float v1 = (mr[16 + lr] != 0) ? s[1][r] : NEGVAL;
      float t = fmaxf(v0, v1);
      t = fmaxf(t, __shfl_xor(t, 1, 16));
      t = fmaxf(t, __shfl_xor(t, 2, 16));
      t = fmaxf(t, __shfl_xor(t, 4, 16));
      t = fmaxf(t, __shfl_xor(t, 8, 16));
      const float mn = fmaxf(rm[r], t);
      const float sc = __expf(rm[r] - mn);
      rm[r] = mn;
      const float p0 = __expf(v0 - mn);
      const float p1 = __expf(v1 - mn);
      float ps = p0 + p1;
      ps += __shfl_xor(ps, 1, 16);
      ps += __shfl_xor(ps, 2, 16);
      ps += __shfl_xor(ps, 4, 16);
      ps += __shfl_xor(ps, 8, 16);
      rl[r] = rl[r] * sc + ps;
#pragma unroll
      for (int j = 0; j < 4; ++j) o[j][r] *= sc;
      // stash P (D-layout -> LDS tile, row = M, col = key-in-tile)
      Plds[wave][hi * 8 + r][lr]      = (bf16_t)p0;
      Plds[wave][hi * 8 + r][16 + lr] = (bf16_t)p1;
    }
    // same-wave LDS store -> cross-lane load ordering
    asm volatile("s_wait_dscnt 0" ::: "memory");

    // ---- regather P as an A-fragment (16x32) ----
    v16bf pa;
#pragma unroll
    for (int i = 0; i < 8; ++i) {
      pa[i]     = Plds[wave][lr][8 * hi + i];
      pa[8 + i] = Plds[wave][lr][16 + 8 * hi + i];
    }

    // ---- O += P * V (V transposed: contiguous along keys) ----
#pragma unroll
    for (int n0 = 0; n0 < 4; ++n0) {
      v16bf bv = load_frag(Vh + (size_t)(n0 * 16 + lr) * L_SEQ + kb, hi);
      o[n0] = wmma_bf16(pa, bv, o[n0]);
    }
  }

  // ---- normalize rows, store O bf16 as [B*L, D] for the output projection
  const int b = bh >> 4, h = bh & 15;
#pragma unroll
  for (int n0 = 0; n0 < 4; ++n0)
#pragma unroll
    for (int r = 0; r < 8; ++r) {
      const int qrow = q0 + hi * 8 + r;
      O[((size_t)(b * L_SEQ + qrow)) * D_MODEL + h * D_HEAD + n0 * 16 + lr] =
          (bf16_t)(o[n0][r] / rl[r]);
    }
}

extern "C" void kernel_launch(void* const* d_in, const int* in_sizes, int n_in,
                              void* d_out, int out_size, void* d_ws, size_t ws_size,
                              hipStream_t stream) {
  (void)in_sizes; (void)n_in; (void)out_size; (void)ws_size;
  const float* query = (const float*)d_in[0];
  const int*   mask  = (const int*)d_in[1];
  const float* Wq = (const float*)d_in[2];
  const float* bq = (const float*)d_in[3];
  const float* Wk = (const float*)d_in[4];
  const float* bk = (const float*)d_in[5];
  const float* Wv = (const float*)d_in[6];
  const float* bv = (const float*)d_in[7];
  const float* Wo = (const float*)d_in[8];
  const float* bo = (const float*)d_in[9];
  float* out = (float*)d_out;

  const size_t M = (size_t)BATCH * L_SEQ;  // 4096
  const size_t D = D_MODEL;                // 1024

  char* ws = (char*)d_ws;                  // ~48 MB bf16 staging
  bf16_t* Xb  = (bf16_t*)ws;               // [M,D]
  bf16_t* Wqb = (bf16_t*)(ws + 2 * M * D); // [D,D] each
  bf16_t* Wkb = Wqb + D * D;
  bf16_t* Wvb = Wkb + D * D;
  bf16_t* Wob = Wvb + D * D;
  bf16_t* Qb  = Wob + D * D;               // [B,H,L,dk]
  bf16_t* Kb  = Qb + M * D;                // [B,H,L,dk]
  bf16_t* Vtb = Kb + M * D;                // [B,H,dk,L]
  bf16_t* Ob  = Vtb + M * D;               // [M,D]

  dim3 blk(256);
  cvt_f32_to_bf16<<<1024, blk, 0, stream>>>(query, Xb, (int)(M * D));
  cvt_f32_to_bf16<<<512,  blk, 0, stream>>>(Wq, Wqb, (int)(D * D));
  cvt_f32_to_bf16<<<512,  blk, 0, stream>>>(Wk, Wkb, (int)(D * D));
  cvt_f32_to_bf16<<<512,  blk, 0, stream>>>(Wv, Wvb, (int)(D * D));
  cvt_f32_to_bf16<<<512,  blk, 0, stream>>>(Wo, Wob, (int)(D * D));

  const int gblocks = (int)((M / 128) * (D / 128));  // 256 blocks, 8 waves each
  gemm_bf16_wmma<<<gblocks, blk, 0, stream>>>(Xb, Wqb, bq, Qb,
                                              (int)M, (int)D, (int)D, 0, 0.125f);
  gemm_bf16_wmma<<<gblocks, blk, 0, stream>>>(Xb, Wkb, bk, Kb,
                                              (int)M, (int)D, (int)D, 0, 1.0f);
  gemm_bf16_wmma<<<gblocks, blk, 0, stream>>>(Xb, Wvb, bv, Vtb,
                                              (int)M, (int)D, (int)D, 1, 1.0f);

  dim3 agrid(L_SEQ / 128, BATCH * N_HEADS);
  attn_wmma<<<agrid, blk, 0, stream>>>(Qb, Kb, Vtb, mask, Ob);

  gemm_bf16_wmma<<<gblocks, blk, 0, stream>>>(Ob, Wob, bo, out,
                                              (int)M, (int)D, (int)D, 2, 1.0f);
}
